// BaseModel_88991722373300
// MI455X (gfx1250) — matrix operand order repaired and checked by reading, not compiled, not run on hardware
//
#include <hip/hip_runtime.h>
#include <hip/hip_bf16.h>

typedef __attribute__((ext_vector_type(16))) _Float16 v16h;
typedef __attribute__((ext_vector_type(8)))  _Float16 v8h;
typedef __attribute__((ext_vector_type(8)))  float    v8f;

// ---------------------------------------------------------------------------
// Stage 1: conv1 7x7 s2 p3 (3->64) + BN + ReLU.  [64,3,32,32] -> [64,64,16,16]
// Direct VALU conv (Cin=3, irregular K=147 -> not worth WMMA re-layout).
// ---------------------------------------------------------------------------
__global__ __launch_bounds__(256) void conv1_bn_relu(
    const float* __restrict__ img, const float* __restrict__ w,
    const float* __restrict__ bs, const float* __restrict__ bb,
    const float* __restrict__ bm, const float* __restrict__ bv,
    float* __restrict__ out)
{
    int idx = blockIdx.x * 256 + threadIdx.x;          // 64*64*16*16 = 1048576
    int ox = idx & 15, oy = (idx >> 4) & 15, co = (idx >> 8) & 63, b = idx >> 14;
    float acc = 0.f;
    for (int ci = 0; ci < 3; ++ci)
        for (int ky = 0; ky < 7; ++ky) {
            int iy = oy * 2 - 3 + ky;
            if ((unsigned)iy >= 32u) continue;
            for (int kx = 0; kx < 7; ++kx) {
                int ix = ox * 2 - 3 + kx;
                if ((unsigned)ix >= 32u) continue;
                acc += img[((b * 3 + ci) * 32 + iy) * 32 + ix] *
                       w[((co * 3 + ci) * 7 + ky) * 7 + kx];
            }
        }
    float iv = bs[co] * rsqrtf(bv[co] + 1e-5f);
    out[idx] = fmaxf(acc * iv + (bb[co] - bm[co] * iv), 0.f);
}

// ---------------------------------------------------------------------------
// Stage 2: maxpool 3x3 s2 p1: [64,64,16,16] -> [64,64,8,8]
// ---------------------------------------------------------------------------
__global__ __launch_bounds__(256) void maxpool3s2(
    const float* __restrict__ in, float* __restrict__ out)
{
    int idx = blockIdx.x * 256 + threadIdx.x;          // 262144
    int x = idx & 7, y = (idx >> 3) & 7, c = (idx >> 6) & 63, b = idx >> 12;
    float mx = -1e30f;
    for (int ky = 0; ky < 3; ++ky) {
        int iy = y * 2 - 1 + ky;
        if ((unsigned)iy >= 16u) continue;
        for (int kx = 0; kx < 3; ++kx) {
            int ix = x * 2 - 1 + kx;
            if ((unsigned)ix >= 16u) continue;
            mx = fmaxf(mx, in[((b * 64 + c) * 16 + iy) * 16 + ix]);
        }
    }
    out[idx] = mx;
}

// ---------------------------------------------------------------------------
// Pack 3x3 conv weights [64,64,3,3] fp32 -> f16, WMMA A fragments, contiguous k order:
// aw[((c*18+kc)*4+t)*32+lane)*16+e], A layout (16x32 f16):
//   lane=16h+m ; K=32kc+16h+e ; A[m][K]=w[co=16t+m][K]
// K = ci*9 + (ky*3+kx)
// ---------------------------------------------------------------------------
__global__ __launch_bounds__(256) void prep_aw(
    const float* __restrict__ w0, const float* __restrict__ w1,
    const float* __restrict__ w2, const float* __restrict__ w3,
    _Float16* __restrict__ aw)
{
    int idx = blockIdx.x * 256 + threadIdx.x;          // 4*18*4*32*16 = 147456
    if (idx >= 147456) return;
    int e    = idx & 15;
    int lane = (idx >> 4) & 31;
    int t    = (idx >> 9) & 3;
    int kc   = (idx >> 11) % 18;
    int c    = (idx >> 11) / 18;
    int h = lane >> 4, m = lane & 15;
    int kl = 16 * h + e;
    int K  = kc * 32 + kl;
    int ci = K / 9, r = K % 9;
    const float* w = (c == 0) ? w0 : (c == 1) ? w1 : (c == 2) ? w2 : w3;
    aw[idx] = (_Float16)w[((t * 16 + m) * 64 + ci) * 9 + r];
}

// elementwise fp32 -> f16 convert (for wdyn_w and v)
__global__ __launch_bounds__(256) void cvt_f32_f16(
    const float* __restrict__ in, _Float16* __restrict__ out, int n)
{
    int i = blockIdx.x * 256 + threadIdx.x;
    if (i < n) out[i] = (_Float16)in[i];
}

// ---------------------------------------------------------------------------
// Stage 3: 3x3 conv 64->64 pad1 + BN (+residual) + ReLU as implicit GEMM.
// One workgroup per batch image, 4 waves; wave t owns co tile [16t,16t+16).
// GEMM: C[64 co x 64 pix] = W[64 x 576] * im2col[576 x 64], K split in two
// 288-wide phases so the LDS im2col matrix (bmat[p][kl], f16) fits in 36 KB.
// ---------------------------------------------------------------------------
__global__ __launch_bounds__(128) void conv3x3_wmma(
    const float* __restrict__ in,      // [64][64][8][8]
    const _Float16* __restrict__ aw,   // packed A fragments, 36864 halves
    const float* __restrict__ bs, const float* __restrict__ bb,
    const float* __restrict__ bm, const float* __restrict__ bv,
    const float* __restrict__ res,     // residual (nullable)
    float* __restrict__ out)
{
    __shared__ float atile[4096];
    __shared__ alignas(32) _Float16 bmat[64 * 288];    // [pixel][k_local]

    const int b    = blockIdx.x;
    const int tid  = threadIdx.x;
    const int wave = tid >> 5;                         // co tile
    const int lane = tid & 31;
    const int h = lane >> 4, ln = lane & 15;

    for (int i = tid; i < 4096; i += 128) atile[i] = in[b * 4096 + i];
    __syncthreads();

    v8f acc[4];
    for (int pt = 0; pt < 4; ++pt)
        for (int j = 0; j < 8; ++j) acc[pt][j] = 0.f;

    for (int phase = 0; phase < 2; ++phase) {
        __syncthreads();                               // previous-phase reads done
        {   // build im2col slab: k in [phase*288, phase*288+288)
            int p  = tid & 63;
            int hs = tid >> 6;                         // interleave 2 threads per pixel
            int y = p >> 3, x = p & 7;
            int kg = phase * 288 + hs;
            int ci = kg / 9, r = kg % 9;
            for (int kl = hs; kl < 288; kl += 2) {
                int ky = r / 3, kx = r - ky * 3;
                int yy = y + ky - 1, xx = x + kx - 1;
                float vv = 0.f;
                if ((unsigned)yy < 8u && (unsigned)xx < 8u)
                    vv = atile[ci * 64 + yy * 8 + xx];
                bmat[p * 288 + kl] = (_Float16)vv;
                r += 2; if (r >= 9) { r -= 9; ci += 1; }
            }
        }
        __syncthreads();
        for (int kc = 0; kc < 9; ++kc) {
            int kcg = phase * 9 + kc;
            // A fragment: one contiguous 32B load per lane
            v16h afr = *(const v16h*)(aw + (((kcg * 4 + wave) * 32 + lane) << 4));
            for (int pt = 0; pt < 4; ++pt) {
                int p = pt * 16 + ln;
                v16h bfr = *(const v16h*)(&bmat[p * 288 + kc * 32 + h * 16]);
                acc[pt] = __builtin_amdgcn_wmma_f32_16x16x32_f16(
                    false, afr, false, bfr, (short)0, acc[pt], false, false);
            }
        }
    }

    // epilogue: D[8h+r][ln] per VGPR r; co = 16*wave + 8h + r
    float inv[8], beta[8];
    for (int r = 0; r < 8; ++r) {
        int co = wave * 16 + h * 8 + r;
        float iv = bs[co] * rsqrtf(bv[co] + 1e-5f);
        inv[r] = iv;
        beta[r] = bb[co] - bm[co] * iv;
    }
    for (int pt = 0; pt < 4; ++pt) {
        int p = pt * 16 + ln;
        for (int r = 0; r < 8; ++r) {
            int co = wave * 16 + h * 8 + r;
            float vl = acc[pt][r] * inv[r] + beta[r];
            if (res) vl += res[b * 4096 + co * 64 + p];
            out[b * 4096 + co * 64 + p] = fmaxf(vl, 0.f);
        }
    }
}

// ---------------------------------------------------------------------------
// Stage 4: w = tanh(v[64,4096] @ wdyn_w.T[4096,576] + b)  via f16 WMMA.
// 36 blocks (N tiles) x 4 waves (M tiles); 128 K-chunks of 32.
// ---------------------------------------------------------------------------
__global__ __launch_bounds__(128) void gemm_dyn(
    const _Float16* __restrict__ vh,   // [64][4096] f16
    const _Float16* __restrict__ wdh,  // [576][4096] f16
    const float* __restrict__ wb,      // [576]
    float* __restrict__ wout)          // [64][576]
{
    int nt = blockIdx.x;               // 0..35
    int mt = threadIdx.x >> 5;         // wave = batch tile
    int lane = threadIdx.x & 31;
    int h = lane >> 4, ln = lane & 15;
    int row = mt * 16 + ln;            // A row (batch)
    int ng  = nt * 16 + ln;            // B col (output feature)

    v8f acc;
    for (int j = 0; j < 8; ++j) acc[j] = 0.f;

    const _Float16* arow = vh  + row * 4096;
    const _Float16* brow = wdh + ng  * 4096;
    for (int kc = 0; kc < 128; ++kc) {
        int kb = kc * 32;
        v8h a0 = *(const v8h*)(arow + kb + 16 * h);
        v8h a1 = *(const v8h*)(arow + kb + 16 * h + 8);
        v16h afr;
        for (int j = 0; j < 8; ++j) { afr[j] = a0[j]; afr[8 + j] = a1[j]; }
        v16h bfr = *(const v16h*)(brow + kb + 16 * h);
        acc = __builtin_amdgcn_wmma_f32_16x16x32_f16(
            false, afr, false, bfr, (short)0, acc, false, false);
    }
    float bias = wb[ng];
    for (int r = 0; r < 8; ++r) {
        int bg = mt * 16 + h * 8 + r;  // global batch row
        wout[bg * 576 + ng] = tanhf(acc[r] + bias);
    }
}

// column L2 norms over the batch dim (axis 0), clamped at 1e-12
__global__ __launch_bounds__(256) void colnorm_k(
    const float* __restrict__ w, float* __restrict__ cn)
{
    int n = blockIdx.x * 256 + threadIdx.x;
    if (n >= 576) return;
    float s = 0.f;
    for (int b = 0; b < 64; ++b) { float x = w[b * 576 + n]; s += x * x; }
    cn[n] = fmaxf(sqrtf(s), 1e-12f);
}

// ---------------------------------------------------------------------------
// Stage 5: per-sample dynamic conv (+dc_b), per-sample L2 normalize, classifier.
// One workgroup per batch sample, 64 threads = 64 output pixels.
// ---------------------------------------------------------------------------
__global__ __launch_bounds__(64) void dynconv_cls(
    const float* __restrict__ vfin, const float* __restrict__ w,
    const float* __restrict__ cn, const float* __restrict__ dcb,
    const float* __restrict__ clsw, const float* __restrict__ clsb,
    float* __restrict__ out)
{
    __shared__ float vt[4096];
    __shared__ float wn[576];
    __shared__ float red[64];
    __shared__ float vhn[64];
    int b = blockIdx.x, p = threadIdx.x;
    for (int i = p; i < 4096; i += 64) vt[i] = vfin[b * 4096 + i];
    for (int k = p; k < 576; k += 64) wn[k] = w[b * 576 + k] / cn[k];
    __syncthreads();
    int y = p >> 3, x = p & 7;
    float acc = dcb[0];
    for (int c = 0; c < 64; ++c)
        for (int ky = 0; ky < 3; ++ky) {
            int yy = y + ky - 1;
            if ((unsigned)yy >= 8u) continue;
            for (int kx = 0; kx < 3; ++kx) {
                int xx = x + kx - 1;
                if ((unsigned)xx >= 8u) continue;
                acc += vt[c * 64 + yy * 8 + xx] * wn[c * 9 + ky * 3 + kx];
            }
        }
    red[p] = acc * acc;
    __syncthreads();
    for (int off = 32; off > 0; off >>= 1) {
        if (p < off) red[p] += red[p + off];
        __syncthreads();
    }
    vhn[p] = acc / sqrtf(red[0]);
    __syncthreads();
    if (p < 10) {
        float s = clsb[p];
        for (int q = 0; q < 64; ++q) s += vhn[q] * clsw[p * 64 + q];
        out[b * 10 + p] = s;
    }
}

// ---------------------------------------------------------------------------
// d_in order (setup_inputs leaves, flattened):
//  0 imgs | 1 conv1_w | 2..5 bn1(s,b,m,v)
//  6 w11 | 7..10 bn11 | 11 w12 | 12..15 bn12      (layer1 block 0)
// 16 w21 | 17..20 bn21 | 21 w22 | 22..25 bn22     (layer1 block 1)
// 26 wdyn_w | 27 wdyn_b | 28 dc_b | 29 wcls_w | 30 wcls_b
// ---------------------------------------------------------------------------
extern "C" void kernel_launch(void* const* d_in, const int* in_sizes, int n_in,
                              void* d_out, int out_size, void* d_ws, size_t ws_size,
                              hipStream_t stream)
{
    (void)in_sizes; (void)n_in; (void)out_size; (void)ws_size;
    const float* F(int i);
    #define FP(i) ((const float*)d_in[(i)])

    char* ws = (char*)d_ws;
    float*    X1   = (float*)(ws + 0);          //  4 MB  [64,64,16,16]
    float*    XP   = (float*)(ws + 4194304);    //  1 MB  pooled / residual
    float*    MID  = (float*)(ws + 5242880);    //  1 MB
    float*    B1O  = (float*)(ws + 6291456);    //  1 MB
    float*    VFIN = (float*)(ws + 7340032);    //  1 MB  layer1 output v
    _Float16* AW   = (_Float16*)(ws + 8388608); //  288 KB packed conv weights
    _Float16* VH   = (_Float16*)(ws + 8683520); //  512 KB v in f16
    _Float16* WDH  = (_Float16*)(ws + 9207808); //  4.5 MB wdyn_w in f16
    float*    WBUF = (float*)(ws + 13926400);   //  144 KB tanh output [64,576]
    float*    CN   = (float*)(ws + 14073856);   //  2.25 KB column norms

    conv1_bn_relu<<<4096, 256, 0, stream>>>(FP(0), FP(1), FP(2), FP(3), FP(4), FP(5), X1);
    maxpool3s2<<<1024, 256, 0, stream>>>(X1, XP);
    prep_aw<<<576, 256, 0, stream>>>(FP(6), FP(11), FP(16), FP(21), AW);
    cvt_f32_f16<<<9216, 256, 0, stream>>>(FP(26), WDH, 2359296);

    conv3x3_wmma<<<64, 128, 0, stream>>>(XP,  AW + 0 * 36864, FP(7),  FP(8),  FP(9),  FP(10), nullptr, MID);
    conv3x3_wmma<<<64, 128, 0, stream>>>(MID, AW + 1 * 36864, FP(12), FP(13), FP(14), FP(15), XP,      B1O);
    conv3x3_wmma<<<64, 128, 0, stream>>>(B1O, AW + 2 * 36864, FP(17), FP(18), FP(19), FP(20), nullptr, MID);
    conv3x3_wmma<<<64, 128, 0, stream>>>(MID, AW + 3 * 36864, FP(22), FP(23), FP(24), FP(25), B1O,     VFIN);

    cvt_f32_f16<<<1024, 256, 0, stream>>>(VFIN, VH, 262144);
    gemm_dyn<<<36, 128, 0, stream>>>(VH, WDH, FP(27), WBUF);
    colnorm_k<<<3, 256, 0, stream>>>(WBUF, CN);
    dynconv_cls<<<64, 64, 0, stream>>>(VFIN, WBUF, CN, FP(28), FP(29), FP(30), (float*)d_out);
    #undef FP
}
